// CausalSelfAttention_3770981286300
// MI455X (gfx1250) — compile-verified, matrix operand order checked by reading, not run on hardware
//
#include <hip/hip_runtime.h>

// Problem constants (match reference)
#define BATCH 4
#define SEQ   2048
#define EMB   1024
#define NHEAD 16
#define HDIM  64
#define M_TOT (BATCH * SEQ)   // 8192
#define N_QKV (3 * EMB)       // 3072

typedef __bf16 bf16;
typedef __attribute__((ext_vector_type(16))) __bf16 v16bf;
typedef __attribute__((ext_vector_type(8)))  float  v8f;

static __device__ __forceinline__ v8f wmma_bf16(v16bf a, v16bf b, v8f c) {
  // D = A(16x32 bf16) * B(32x16 bf16) + C(16x16 f32)
  return __builtin_amdgcn_wmma_f32_16x16x32_bf16(false, a, false, b, (short)0, c,
                                                 false, false);
}

// Fragment load from LDS, contiguous per lane -> 2x ds_load_b128.
// A-frag from [m][k] storage; B-frag from N-major [n][k] storage.
static __device__ __forceinline__ v16bf load_frag(const bf16* p, int stride,
                                                  int r0, int k0, int lane) {
  int r  = r0 + (lane & 15);
  int kh = k0 + ((lane >> 4) << 3);
  const bf16* row = p + (size_t)r * stride + kh;
  v16bf a;
#pragma unroll
  for (int e = 0; e < 8; ++e) {
    a[e]     = row[e];
    a[e + 8] = row[16 + e];
  }
  return a;
}

static __device__ __forceinline__ void pack4_store(bf16* dst, float x, float y,
                                                   float z, float w) {
  bf16 tmp[4] = {(bf16)x, (bf16)y, (bf16)z, (bf16)w};
  *(unsigned long long*)dst = *(const unsigned long long*)tmp;  // ds_store_b64
}

// CDNA5 async global->LDS DMA, 16 bytes per lane, tracked by ASYNCcnt.
// LDS byte address = low 32 bits of the flat shared-pointer (aperture rules).
static __device__ __forceinline__ void async_copy_b128(void* lds_dst,
                                                       const void* gsrc) {
  unsigned lds = (unsigned)(unsigned long long)lds_dst;
  unsigned long long g = (unsigned long long)gsrc;
  asm volatile("global_load_async_to_lds_b128 %0, %1, off"
               :: "v"(lds), "v"(g) : "memory");
}
static __device__ __forceinline__ void wait_async0() {
  asm volatile("s_wait_asynccnt 0x0" ::: "memory");
}

// ---------------------------------------------------------------------------
// Kernel 1: qkv = x @ w_qkv + b_qkv, scattered into Q/K/V [B,H,T,D] bf16.
// 256 thr = 8 waves; block tile 128(M) x 128(N); wave tile 32 x 64; K-step 32.
// Double-buffered LDS; fp32->bf16 convert in staging.
// ---------------------------------------------------------------------------
__global__ __launch_bounds__(256)
void qkv_gemm_kernel(const float* __restrict__ X, const float* __restrict__ W,
                     const float* __restrict__ bias, bf16* __restrict__ Q,
                     bf16* __restrict__ Kt, bf16* __restrict__ V) {
  __shared__ bf16 As[2][128][40];    // [m][k]
  __shared__ bf16 Bst[2][128][40];   // [n][k]  (N-major => contiguous frags)

  const int tid  = threadIdx.x;
  const int lane = tid & 31;
  const int wave = tid >> 5;
  const int wm   = (wave >> 1) * 32;
  const int wn   = (wave & 1) * 64;
  const int m0   = blockIdx.y * 128;
  const int n0   = blockIdx.x * 128;
  const int nB   = tid & 127;        // B staging column
  const int kB   = (tid >> 7) * 16;  // B staging k half

  v8f acc[2][4];
#pragma unroll
  for (int ti = 0; ti < 2; ++ti)
#pragma unroll
    for (int tj = 0; tj < 4; ++tj) acc[ti][tj] = v8f{};

  // staging: fixed-trip, load-phase then store-phase (branchless, pipelined)
  auto stage = [&](int buf, int k0) {
    float4 fa[4];
#pragma unroll
    for (int it = 0; it < 4; ++it) {
      int idx = tid * 4 + it * 1024;  // A tile 128x32 fp32
      int r = idx >> 5, c = idx & 31;
      fa[it] = *(const float4*)&X[(size_t)(m0 + r) * EMB + k0 + c];
    }
    float fb[16];
#pragma unroll
    for (int u = 0; u < 16; ++u)      // B tile 32x128 fp32 (transposed store)
      fb[u] = W[(size_t)(k0 + kB + u) * N_QKV + n0 + nB];
#pragma unroll
    for (int it = 0; it < 4; ++it) {
      int idx = tid * 4 + it * 1024;
      int r = idx >> 5, c = idx & 31;
      pack4_store(&As[buf][r][c], fa[it].x, fa[it].y, fa[it].z, fa[it].w);
    }
#pragma unroll
    for (int j = 0; j < 4; ++j)
      pack4_store(&Bst[buf][nB][kB + 4 * j], fb[4 * j], fb[4 * j + 1],
                  fb[4 * j + 2], fb[4 * j + 3]);
  };

  stage(0, 0);
  for (int kt = 0; kt < EMB / 32; ++kt) {
    __syncthreads();
    if (kt + 1 < EMB / 32) stage((kt + 1) & 1, (kt + 1) * 32);
    const int cur = kt & 1;
    v16bf a0 = load_frag(&As[cur][0][0], 40, wm, 0, lane);
    v16bf a1 = load_frag(&As[cur][0][0], 40, wm + 16, 0, lane);
#pragma unroll
    for (int tj = 0; tj < 4; ++tj) {
      v16bf b = load_frag(&Bst[cur][0][0], 40, wn + tj * 16, 0, lane);
      acc[0][tj] = wmma_bf16(a0, b, acc[0][tj]);
      acc[1][tj] = wmma_bf16(a1, b, acc[1][tj]);
    }
  }

  // epilogue: add bias, scatter n -> (s, h, d), m -> (b, t) into [B,H,T,D]
  const int hl = lane >> 4;
  const int nc = lane & 15;
  const int s  = n0 >> 10;  // 128-col block never crosses a 1024 boundary
  bf16* outp = (s == 0) ? Q : ((s == 1) ? Kt : V);
#pragma unroll
  for (int ti = 0; ti < 2; ++ti) {
#pragma unroll
    for (int tj = 0; tj < 4; ++tj) {
#pragma unroll
      for (int i = 0; i < 8; ++i) {
        int m = m0 + wm + ti * 16 + i + 8 * hl;
        int n = n0 + wn + tj * 16 + nc;
        float v = acc[ti][tj][i] + bias[n];
        int h = (n >> 6) & (NHEAD - 1);
        int d = n & (HDIM - 1);
        int b = m >> 11;       // / SEQ
        int t = m & (SEQ - 1);
        outp[(((size_t)b * NHEAD + h) * SEQ + t) * HDIM + d] = (bf16)v;
      }
    }
  }
}

// ---------------------------------------------------------------------------
// Kernel 2: flash attention. 128 thr = 4 waves = 64 queries of one (b,h).
// Q/K tiles staged with async global->LDS DMA (ASYNCcnt).
// ---------------------------------------------------------------------------
__global__ __launch_bounds__(128)
void attn_kernel(const bf16* __restrict__ Q, const bf16* __restrict__ K,
                 const bf16* __restrict__ V, bf16* __restrict__ Y) {
  __shared__ bf16 Qs[64][72];    // [q][d]
  __shared__ bf16 Ksh[64][72];   // [key][d]
  __shared__ bf16 Vst[64][72];   // [d][key]  (transposed => contiguous frags)
  __shared__ bf16 Ps[4][16][72]; // per-wave P round-trip

  const int tid  = threadIdx.x;
  const int lane = tid & 31;
  const int wave = tid >> 5;
  const int bh   = blockIdx.y;
  const int q0   = blockIdx.x * 64;
  const int b    = bh / NHEAD;
  const int h    = bh % NHEAD;
  const size_t base = (size_t)bh * SEQ * HDIM;
  const float scale = 0.125f;  // 1/sqrt(64)

  // stage Q via async DMA (4x 16B per thread)
#pragma unroll
  for (int it = 0; it < 4; ++it) {
    int idx = tid + it * 128;
    int r = idx >> 3, c8 = (idx & 7) * 8;
    async_copy_b128(&Qs[r][c8], &Q[base + (size_t)(q0 + r) * HDIM + c8]);
  }
  wait_async0();
  __syncthreads();

  const v16bf qa0 = load_frag(&Qs[0][0], 72, wave * 16, 0,  lane);
  const v16bf qa1 = load_frag(&Qs[0][0], 72, wave * 16, 32, lane);

  v8f oacc[4] = {v8f{}, v8f{}, v8f{}, v8f{}};
  float mrow[8], lrow[8];
#pragma unroll
  for (int i = 0; i < 8; ++i) { mrow[i] = -1e30f; lrow[i] = 0.0f; }

  const int hl  = lane >> 4;
  const int nc  = lane & 15;
  const int dV  = tid & 63;          // V transpose staging: d column
  const int kV  = (tid >> 6) * 32;   //                      key half
  const int nkb = blockIdx.x + 1;    // causal: key blocks 0 .. q-tile

  for (int kbi = 0; kbi < nkb; ++kbi) {
    const int kb = kbi * 64;
    // stage K rows via async DMA
#pragma unroll
    for (int it = 0; it < 4; ++it) {
      int idx = tid + it * 128;
      int r = idx >> 3, c8 = (idx & 7) * 8;
      async_copy_b128(&Ksh[r][c8], &K[base + (size_t)(kb + r) * HDIM + c8]);
    }
    // stage V transposed: Vst[d][key] (gather + packed ds_store_b64)
    bf16 tv[32];
#pragma unroll
    for (int u = 0; u < 32; ++u)
      tv[u] = V[base + (size_t)(kb + kV + u) * HDIM + dV];
#pragma unroll
    for (int j = 0; j < 8; ++j)
      *(unsigned long long*)&Vst[dV][kV + 4 * j] =
          *(const unsigned long long*)&tv[4 * j];
    wait_async0();
    __syncthreads();

    // S = Q K^T
    v8f sacc[4] = {v8f{}, v8f{}, v8f{}, v8f{}};
#pragma unroll
    for (int tj = 0; tj < 4; ++tj) {
      v16bf bk0 = load_frag(&Ksh[0][0], 72, tj * 16, 0,  lane);
      v16bf bk1 = load_frag(&Ksh[0][0], 72, tj * 16, 32, lane);
      sacc[tj] = wmma_bf16(qa0, bk0, sacc[tj]);
      sacc[tj] = wmma_bf16(qa1, bk1, sacc[tj]);
    }

    // online softmax (row stats per accumulator register i)
    const bool diag = (kb + 63 > q0);
#pragma unroll
    for (int i = 0; i < 8; ++i) {
      const int qg = q0 + wave * 16 + i + 8 * hl;
      float sv[4];
      float rmax = -1e30f;
#pragma unroll
      for (int tj = 0; tj < 4; ++tj) {
        float v = sacc[tj][i] * scale;
        if (diag && (kb + tj * 16 + nc) > qg) v = -1e30f;
        sv[tj] = v;
        rmax = fmaxf(rmax, v);
      }
#pragma unroll
      for (int mk = 1; mk < 16; mk <<= 1)  // half-wave (row) reduction
        rmax = fmaxf(rmax, __shfl_xor(rmax, mk, 32));
      const float newm  = fmaxf(mrow[i], rmax);
      const float alpha = __expf(mrow[i] - newm);
      float rsum = 0.0f;
#pragma unroll
      for (int tj = 0; tj < 4; ++tj) {
        float p = __expf(sv[tj] - newm);
        sv[tj] = p;
        rsum += p;
      }
#pragma unroll
      for (int mk = 1; mk < 16; mk <<= 1)
        rsum += __shfl_xor(rsum, mk, 32);
      lrow[i] = lrow[i] * alpha + rsum;
      mrow[i] = newm;
#pragma unroll
      for (int tj = 0; tj < 4; ++tj) oacc[tj][i] *= alpha;
      const int r = i + 8 * hl;
#pragma unroll
      for (int tj = 0; tj < 4; ++tj)
        Ps[wave][r][tj * 16 + nc] = (bf16)sv[tj];
    }
    asm volatile("s_wait_dscnt 0x0" ::: "memory");  // wave-private P round-trip

    // O += P V   (V B-frags are contiguous reads from Vst[d][key])
    v16bf pa0 = load_frag(&Ps[wave][0][0], 72, 0, 0,  lane);
    v16bf pa1 = load_frag(&Ps[wave][0][0], 72, 0, 32, lane);
#pragma unroll
    for (int tj = 0; tj < 4; ++tj) {
      v16bf vb0 = load_frag(&Vst[0][0], 72, tj * 16, 0,  lane);
      v16bf vb1 = load_frag(&Vst[0][0], 72, tj * 16, 32, lane);
      oacc[tj] = wmma_bf16(pa0, vb0, oacc[tj]);
      oacc[tj] = wmma_bf16(pa1, vb1, oacc[tj]);
    }
    __syncthreads();
  }

  // y[b, t, h*64 + d] = O / l
#pragma unroll
  for (int tj = 0; tj < 4; ++tj) {
#pragma unroll
    for (int i = 0; i < 8; ++i) {
      const int t = q0 + wave * 16 + i + 8 * hl;
      const int c = h * HDIM + tj * 16 + nc;
      const float v = oacc[tj][i] / lrow[i];
      Y[((size_t)b * SEQ + t) * EMB + c] = (bf16)v;
    }
  }
}

// ---------------------------------------------------------------------------
// Kernel 3: out = y @ w_out + b_out (fp32 result). Double-buffered;
// A tile staged with async global->LDS DMA (bf16 copy, no convert needed).
// ---------------------------------------------------------------------------
__global__ __launch_bounds__(256)
void out_gemm_kernel(const bf16* __restrict__ Yin, const float* __restrict__ W,
                     const float* __restrict__ bias, float* __restrict__ O) {
  __shared__ bf16 As[2][128][40];    // [m][k]
  __shared__ bf16 Bst[2][128][40];   // [n][k]

  const int tid  = threadIdx.x;
  const int lane = tid & 31;
  const int wave = tid >> 5;
  const int wm   = (wave >> 1) * 32;
  const int wn   = (wave & 1) * 64;
  const int m0   = blockIdx.y * 128;
  const int n0   = blockIdx.x * 128;
  const int nB   = tid & 127;
  const int kB   = (tid >> 7) * 16;

  v8f acc[2][4];
#pragma unroll
  for (int ti = 0; ti < 2; ++ti)
#pragma unroll
    for (int tj = 0; tj < 4; ++tj) acc[ti][tj] = v8f{};

  auto stage = [&](int buf, int k0) {
    // A tile 128x32 bf16: 2x async 16B DMA per thread
#pragma unroll
    for (int it = 0; it < 2; ++it) {
      int idx = tid + it * 256;
      int r = idx >> 2, c8 = (idx & 3) * 8;
      async_copy_b128(&As[buf][r][c8],
                      &Yin[(size_t)(m0 + r) * EMB + k0 + c8]);
    }
    // B tile 32x128 fp32 -> bf16 transposed
    float fb[16];
#pragma unroll
    for (int u = 0; u < 16; ++u)
      fb[u] = W[(size_t)(k0 + kB + u) * EMB + n0 + nB];
#pragma unroll
    for (int j = 0; j < 4; ++j)
      pack4_store(&Bst[buf][nB][kB + 4 * j], fb[4 * j], fb[4 * j + 1],
                  fb[4 * j + 2], fb[4 * j + 3]);
  };

  stage(0, 0);
  for (int kt = 0; kt < EMB / 32; ++kt) {
    wait_async0();
    __syncthreads();
    if (kt + 1 < EMB / 32) stage((kt + 1) & 1, (kt + 1) * 32);
    const int cur = kt & 1;
    v16bf a0 = load_frag(&As[cur][0][0], 40, wm, 0, lane);
    v16bf a1 = load_frag(&As[cur][0][0], 40, wm + 16, 0, lane);
#pragma unroll
    for (int tj = 0; tj < 4; ++tj) {
      v16bf bfr = load_frag(&Bst[cur][0][0], 40, wn + tj * 16, 0, lane);
      acc[0][tj] = wmma_bf16(a0, bfr, acc[0][tj]);
      acc[1][tj] = wmma_bf16(a1, bfr, acc[1][tj]);
    }
  }

  const int hl = lane >> 4;
  const int nc = lane & 15;
#pragma unroll
  for (int ti = 0; ti < 2; ++ti) {
#pragma unroll
    for (int tj = 0; tj < 4; ++tj) {
#pragma unroll
      for (int i = 0; i < 8; ++i) {
        int m = m0 + wm + ti * 16 + i + 8 * hl;
        int n = n0 + wn + tj * 16 + nc;
        O[(size_t)m * EMB + n] = acc[ti][tj][i] + bias[n];
      }
    }
  }
}

// ---------------------------------------------------------------------------
extern "C" void kernel_launch(void* const* d_in, const int* in_sizes, int n_in,
                              void* d_out, int out_size, void* d_ws, size_t ws_size,
                              hipStream_t stream) {
  const float* x     = (const float*)d_in[0];
  const float* w_qkv = (const float*)d_in[1];
  const float* b_qkv = (const float*)d_in[2];
  const float* w_out = (const float*)d_in[3];
  const float* b_out = (const float*)d_in[4];
  float* out = (float*)d_out;

  const size_t per = (size_t)BATCH * NHEAD * SEQ * HDIM;  // 8,388,608 elems
  bf16* Q = (bf16*)d_ws;
  bf16* K = Q + per;
  bf16* V = K + per;
  bf16* Y = V + per;   // [B, T, C] bf16

  qkv_gemm_kernel<<<dim3(N_QKV / 128, M_TOT / 128), dim3(256), 0, stream>>>(
      x, w_qkv, b_qkv, Q, K, V);
  attn_kernel<<<dim3(SEQ / 64, BATCH * NHEAD), dim3(128), 0, stream>>>(
      Q, K, V, Y);
  out_gemm_kernel<<<dim3(EMB / 128, M_TOT / 128), dim3(256), 0, stream>>>(
      Y, w_out, b_out, out);
}
